// QuantAttention_197568495983
// MI455X (gfx1250) — 8-bit matrix operand order repaired and checked by reading, compiled, not run on hardware
//
#include <hip/hip_runtime.h>
#include <hip/hip_bf16.h>

typedef __attribute__((ext_vector_type(8))) int v8i;
typedef __attribute__((ext_vector_type(4))) int v4i;
typedef __attribute__((ext_vector_type(2))) int v2i;

#define BATCH   2
#define NTOK    2048
#define C_DIM   384
#define NHEAD   6
#define HDIM    64
#define M_ROWS  (BATCH * NTOK)
#define QKV_N   (3 * C_DIM)

__device__ __forceinline__ int quant_code(float x, float inv_alpha) {
    float s = fminf(fmaxf(x * inv_alpha, -128.0f), 127.0f);
    return (int)rintf(s);
}

__device__ __forceinline__ v8i load_a8(const signed char* base, int ld, int lane) {
    const signed char* p = base + (lane & 15) * ld + ((lane >> 4) * 16);
    v2i d0 = *(const v2i*)(p);
    v2i d1 = *(const v2i*)(p + 8);
    v2i d2 = *(const v2i*)(p + 32);
    v2i d3 = *(const v2i*)(p + 40);
    v8i a;
    a[0] = d0[0]; a[1] = d0[1]; a[2] = d1[0]; a[3] = d1[1];
    a[4] = d2[0]; a[5] = d2[1]; a[6] = d3[0]; a[7] = d3[1];
    return a;
}

__device__ __forceinline__ v8i load_b8(const signed char* base, int ld, int lane) {
    const signed char* p = base + (lane & 15) * ld + ((lane >> 4) * 16);
    v4i q0 = *(const v4i*)(p);
    v4i q1 = *(const v4i*)(p + 32);
    v8i b;
    b[0] = q0[0]; b[1] = q0[1]; b[2] = q0[2]; b[3] = q0[3];
    b[4] = q1[0]; b[5] = q1[1]; b[6] = q1[2]; b[7] = q1[3];
    return b;
}

__device__ __forceinline__ void stage_k_async(const signed char* src,
                                              signed char* dst_lds, int tid) {
    unsigned      lo = (unsigned)(uintptr_t)dst_lds + (unsigned)(tid * 16);
    unsigned long long ga = (unsigned long long)(uintptr_t)src + (unsigned)(tid * 16);
    asm volatile("global_load_async_to_lds_b128 %0, %1, off"
                 :: "v"(lo), "v"(ga) : "memory");
    asm volatile("global_load_async_to_lds_b128 %0, %1, off offset:2048"
                 :: "v"(lo), "v"(ga) : "memory");
}
__device__ __forceinline__ void wait_async0() {
    asm volatile("s_wait_asynccnt 0" ::: "memory");
}

__global__ void quant_f32_to_i8_kernel(const float* __restrict__ in,
                                       signed char* __restrict__ out,
                                       int n, const float* __restrict__ alpha) {
    float inv = 1.0f / alpha[0];
    int i = blockIdx.x * blockDim.x + threadIdx.x;
    int stride = gridDim.x * blockDim.x;
    for (; i < n; i += stride)
        out[i] = (signed char)quant_code(in[i], inv);
}

__global__ __launch_bounds__(128) void qkv_gemm_kernel(
    const signed char* __restrict__ x8, const signed char* __restrict__ w8,
    signed char* __restrict__ q8o, signed char* __restrict__ k8o,
    signed char* __restrict__ vt8,
    const float* a_qkv_a, const float* a_qkv_w,
    const float* a_q, const float* a_k, const float* a_v) {
    const int lane = threadIdx.x & 31;
    const int wave = threadIdx.x >> 5;
    const int nblk = QKV_N / 64;
    const int mt = blockIdx.x / nblk;
    const int ng = blockIdx.x % nblk;
    const int m0 = mt * 16;
    const int n0 = (ng * 4 + wave) * 16;

    const signed char* Abase = x8 + (long)m0 * C_DIM;
    const signed char* Bbase = w8 + (long)n0 * C_DIM;

    v8i acc = {0, 0, 0, 0, 0, 0, 0, 0};
    #pragma unroll
    for (int k0 = 0; k0 < C_DIM; k0 += 64) {
        v8i a = load_a8(Abase + k0, C_DIM, lane);
        v8i b = load_b8(Bbase + k0, C_DIM, lane);
        acc = __builtin_amdgcn_wmma_i32_16x16x64_iu8(true, a, true, b, acc, false, false);
    }

    const float sc = a_qkv_a[0] * a_qkv_w[0];
    const float iq = 1.0f / a_q[0];
    const float ik = 1.0f / a_k[0];
    const float iv = 1.0f / a_v[0];

    #pragma unroll
    for (int j = 0; j < 8; ++j) {
        int m = m0 + j + 8 * (lane >> 4);
        int n = n0 + (lane & 15);
        float f = (float)acc[j] * sc;
        int which = n / C_DIM;
        int rem   = n % C_DIM;
        int h = rem / HDIM, d = rem % HDIM;
        int b_ = m / NTOK, tok = m % NTOK;
        long bh = (long)(b_ * NHEAD + h);
        if (which == 0)
            q8o[bh * NTOK * HDIM + (long)tok * HDIM + d] = (signed char)quant_code(f, iq);
        else if (which == 1)
            k8o[bh * NTOK * HDIM + (long)tok * HDIM + d] = (signed char)quant_code(f, ik);
        else
            vt8[bh * (long)HDIM * NTOK + (long)d * NTOK + tok] = (signed char)quant_code(f, iv);
    }
}

__global__ __launch_bounds__(128) void attn_kernel(
    const signed char* __restrict__ qg, const signed char* __restrict__ kg,
    const signed char* __restrict__ vtg, signed char* __restrict__ o8,
    const float* a_q, const float* a_k, const float* a_attn,
    const float* a_attn2, const float* a_v, const float* a_proj_a) {
    __shared__ __align__(16) signed char kstage[2][64 * 64];
    __shared__ __align__(16) signed char pbuf[4][16 * 64];

    const int tid  = threadIdx.x;
    const int lane = tid & 31;
    const int wave = tid >> 5;
    const int blocks_per_bh = NTOK / 64;
    const int bh = blockIdx.x / blocks_per_bh;
    const int m0 = (blockIdx.x % blocks_per_bh) * 64 + wave * 16;
    const int NG = NTOK / 64;

    const signed char* Q  = qg  + (long)bh * NTOK * HDIM + (long)m0 * HDIM;
    const signed char* K  = kg  + (long)bh * NTOK * HDIM;
    const signed char* VT = vtg + (long)bh * HDIM * NTOK;

    const float sqk   = a_q[0] * a_k[0] * 0.125f;
    const float aA    = a_attn[0];
    const float invA  = 1.0f / aA;
    const float invA2 = 1.0f / a_attn2[0];
    const float sPV   = a_attn2[0] * a_v[0];
    const float invP  = 1.0f / a_proj_a[0];

    const v8i qa = load_a8(Q, HDIM, lane);

    float rmax[8], rsum[8];
    #pragma unroll
    for (int j = 0; j < 8; ++j) { rmax[j] = -3.0e38f; rsum[j] = 0.0f; }

    const v8i zero = {0, 0, 0, 0, 0, 0, 0, 0};

    stage_k_async(K, kstage[0], tid);
    for (int g = 0; g < NG; ++g) {
        wait_async0();
        __syncthreads();
        if (g + 1 < NG)
            stage_k_async(K + (long)(g + 1) * 64 * HDIM, kstage[(g + 1) & 1], tid);
        const signed char* kb = kstage[g & 1];
        #pragma unroll
        for (int sub = 0; sub < 4; ++sub) {
            v8i b = load_b8(kb + sub * 16 * HDIM, HDIM, lane);
            v8i c = __builtin_amdgcn_wmma_i32_16x16x64_iu8(true, qa, true, b, zero, false, false);
            #pragma unroll
            for (int j = 0; j < 8; ++j) {
                float s  = (float)c[j] * sqk;
                float sq = rintf(fminf(fmaxf(s * invA, -128.0f), 127.0f)) * aA;
                float tm = sq;
                #pragma unroll
                for (int off = 8; off >= 1; off >>= 1)
                    tm = fmaxf(tm, __shfl_xor(tm, off, 32));
                float mnew = fmaxf(rmax[j], tm);
                float e = __expf(sq - mnew);
                #pragma unroll
                for (int off = 8; off >= 1; off >>= 1)
                    e += __shfl_xor(e, off, 32);
                rsum[j] = rsum[j] * __expf(rmax[j] - mnew) + e;
                rmax[j] = mnew;
            }
        }
    }
    float rinv[8];
    #pragma unroll
    for (int j = 0; j < 8; ++j) rinv[j] = 1.0f / rsum[j];

    v8i accd[4];
    #pragma unroll
    for (int dt = 0; dt < 4; ++dt) accd[dt] = zero;

    __syncthreads();
    stage_k_async(K, kstage[0], tid);
    for (int g = 0; g < NG; ++g) {
        wait_async0();
        __syncthreads();
        if (g + 1 < NG)
            stage_k_async(K + (long)(g + 1) * 64 * HDIM, kstage[(g + 1) & 1], tid);
        if (g + 1 < NG)
            __builtin_prefetch(VT + (long)(lane & 3) * 16 * NTOK + (g + 1) * 64, 0, 1);
        const signed char* kb = kstage[g & 1];
        #pragma unroll
        for (int sub = 0; sub < 4; ++sub) {
            v8i b = load_b8(kb + sub * 16 * HDIM, HDIM, lane);
            v8i c = __builtin_amdgcn_wmma_i32_16x16x64_iu8(true, qa, true, b, zero, false, false);
            #pragma unroll
            for (int j = 0; j < 8; ++j) {
                float s  = (float)c[j] * sqk;
                float sq = rintf(fminf(fmaxf(s * invA, -128.0f), 127.0f)) * aA;
                float p  = __expf(sq - rmax[j]) * rinv[j];
                int  pi  = (int)rintf(fminf(fmaxf(p * invA2, -128.0f), 127.0f));
                int  r   = j + 8 * (lane >> 4);
                pbuf[wave][r * 64 + sub * 16 + (lane & 15)] = (signed char)pi;
            }
        }
        v8i pa = load_a8(&pbuf[wave][0], 64, lane);
        #pragma unroll
        for (int dt = 0; dt < 4; ++dt) {
            v8i bv = load_b8(VT + (long)dt * 16 * NTOK + (long)g * 64, NTOK, lane);
            accd[dt] = __builtin_amdgcn_wmma_i32_16x16x64_iu8(false, pa, true, bv,
                                                              accd[dt], false, false);
        }
    }

    const int b_ = bh / NHEAD, h = bh % NHEAD;
    #pragma unroll
    for (int dt = 0; dt < 4; ++dt) {
        #pragma unroll
        for (int j = 0; j < 8; ++j) {
            int tok = m0 + j + 8 * (lane >> 4);
            int col = h * HDIM + dt * 16 + (lane & 15);
            float f = (float)accd[dt][j] * sPV;
            o8[((long)(b_ * NTOK + tok)) * C_DIM + col] = (signed char)quant_code(f, invP);
        }
    }
}

__global__ __launch_bounds__(128) void proj_gemm_kernel(
    const signed char* __restrict__ o8, const signed char* __restrict__ w8,
    const float* __restrict__ bproj, float* __restrict__ out,
    const float* a_a, const float* a_w) {
    const int lane = threadIdx.x & 31;
    const int wave = threadIdx.x >> 5;
    const int nblk = C_DIM / 64;
    const int mt = blockIdx.x / nblk;
    const int ng = blockIdx.x % nblk;
    const int m0 = mt * 16;
    const int n0 = (ng * 4 + wave) * 16;

    const signed char* Abase = o8 + (long)m0 * C_DIM;
    const signed char* Bbase = w8 + (long)n0 * C_DIM;

    v8i acc = {0, 0, 0, 0, 0, 0, 0, 0};
    #pragma unroll
    for (int k0 = 0; k0 < C_DIM; k0 += 64) {
        v8i a = load_a8(Abase + k0, C_DIM, lane);
        v8i b = load_b8(Bbase + k0, C_DIM, lane);
        acc = __builtin_amdgcn_wmma_i32_16x16x64_iu8(true, a, true, b, acc, false, false);
    }

    const float sc = a_a[0] * a_w[0];
    #pragma unroll
    for (int j = 0; j < 8; ++j) {
        int m = m0 + j + 8 * (lane >> 4);
        int n = n0 + (lane & 15);
        out[(long)m * C_DIM + n] = (float)acc[j] * sc + bproj[n];
    }
}

extern "C" void kernel_launch(void* const* d_in, const int* in_sizes, int n_in,
                              void* d_out, int out_size, void* d_ws, size_t ws_size,
                              hipStream_t stream) {
    const float* x      = (const float*)d_in[0];
    const float* w_qkv  = (const float*)d_in[1];
    const float* w_proj = (const float*)d_in[2];
    const float* b_proj = (const float*)d_in[3];
    const float* a_qkv_w  = (const float*)d_in[4];
    const float* a_qkv_a  = (const float*)d_in[5];
    const float* a_proj_w = (const float*)d_in[6];
    const float* a_proj_a = (const float*)d_in[7];
    const float* a_q     = (const float*)d_in[8];
    const float* a_k     = (const float*)d_in[9];
    const float* a_v     = (const float*)d_in[10];
    const float* a_attn  = (const float*)d_in[11];
    const float* a_attn2 = (const float*)d_in[12];
    float* out = (float*)d_out;

    char* ws = (char*)d_ws;
    const size_t SZ_X  = (size_t)M_ROWS * C_DIM;
    const size_t SZ_WQ = (size_t)QKV_N * C_DIM;
    const size_t SZ_WP = (size_t)C_DIM * C_DIM;
    const size_t SZ_HD = (size_t)BATCH * NHEAD * NTOK * HDIM;
    signed char* x_i8  = (signed char*)(ws);
    signed char* wq_i8 = (signed char*)(ws + SZ_X);
    signed char* wp_i8 = (signed char*)(ws + SZ_X + SZ_WQ);
    signed char* q_i8  = (signed char*)(ws + SZ_X + SZ_WQ + SZ_WP);
    signed char* k_i8  = (signed char*)(ws + SZ_X + SZ_WQ + SZ_WP + SZ_HD);
    signed char* vt_i8 = (signed char*)(ws + SZ_X + SZ_WQ + SZ_WP + 2 * SZ_HD);
    signed char* o_i8  = (signed char*)(ws + SZ_X + SZ_WQ + SZ_WP + 3 * SZ_HD);

    quant_f32_to_i8_kernel<<<1024, 256, 0, stream>>>(x, x_i8, (int)SZ_X, a_qkv_a);
    quant_f32_to_i8_kernel<<<512, 256, 0, stream>>>(w_qkv, wq_i8, (int)SZ_WQ, a_qkv_w);
    quant_f32_to_i8_kernel<<<256, 256, 0, stream>>>(w_proj, wp_i8, (int)SZ_WP, a_proj_w);

    qkv_gemm_kernel<<<(M_ROWS / 16) * (QKV_N / 64), 128, 0, stream>>>(
        x_i8, wq_i8, q_i8, k_i8, vt_i8, a_qkv_a, a_qkv_w, a_q, a_k, a_v);

    attn_kernel<<<BATCH * NHEAD * (NTOK / 64), 128, 0, stream>>>(
        q_i8, k_i8, vt_i8, o_i8, a_q, a_k, a_attn, a_attn2, a_v, a_proj_a);

    proj_gemm_kernel<<<(M_ROWS / 16) * (C_DIM / 64), 128, 0, stream>>>(
        o_i8, wp_i8, b_proj, out, a_proj_a, a_proj_w);
}
